// NonIsotropicNCB_29274497089743
// MI455X (gfx1250) — matrix operand order repaired and checked by reading, not compiled, not run on hardware
//
#include <hip/hip_runtime.h>

typedef __bf16 bf16;
typedef __attribute__((ext_vector_type(16))) __bf16 v16bf;
typedef __attribute__((ext_vector_type(8)))  __bf16 v8bf;
typedef __attribute__((ext_vector_type(8)))  float  v8f;

#define SDIM 20
#define S2   400
#define S3   8000
#define SP   160000

// Per-kflat gather descriptor, stored as int2: .x = linear spatial displacement,
// .y = four signed per-dimension offsets packed as bytes (boundary clipping).

// ---------------------------------------------------------------- helpers ---

__global__ __launch_bounds__(256) void cvt_f32_to_bf16(const float* __restrict__ in,
                                                       bf16* __restrict__ out, int n) {
  int i = blockIdx.x * blockDim.x + threadIdx.x;
  if (i < n) out[i] = (bf16)in[i];
}

__global__ __launch_bounds__(64) void zero_bf16(bf16* __restrict__ p, int n) {
  int i = blockIdx.x * blockDim.x + threadIdx.x;
  if (i < n) p[i] = (bf16)0.f;
}

// Build the per-kflat offset table for one pass (decomposition dims d1..d4).
__global__ __launch_bounds__(256) void build_ktab(
    int2* __restrict__ tab, int maxKf,
    int d1, int d2, int d3, int d4) {
  int kf = blockIdx.x * blockDim.x + threadIdx.x;
  if (kf >= maxKf) return;
  int d34 = d3 * d4, d234 = d2 * d34;
  int o1 = kf / d234; int rem = kf - o1 * d234;
  int o2 = rem / d34; rem -= o2 * d34;
  int o3 = rem / d4;  int o4 = rem - o3 * d4;
  int f1 = o1 - ((d1 - 1) >> 1);
  int f2 = o2 - ((d2 - 1) >> 1);
  int f3 = o3 - ((d3 - 1) >> 1);
  int f4 = o4 - ((d4 - 1) >> 1);
  int2 e;
  e.x = f1 * S3 + f2 * S2 + f3 * SDIM + f4;
  e.y = (f1 & 255) | ((f2 & 255) << 8) | ((f3 & 255) << 16) | ((f4 & 255) << 24);
  tab[kf] = e;
}

// Repack weights into the per-lane WMMA B-operand layout:
//   pack[c*512 + q*16 + j]  <->  (k = c*32 + (q>>4)*16 + j, co = q&15)
__global__ __launch_bounds__(256) void repack_weights(
    const float* __restrict__ w, bf16* __restrict__ pack,
    int Cout, int Cin, int KV,
    int d4, int d34, int d234,
    int s1, int s2, int s3, int s4, int nChunks) {
  int idx = blockIdx.x * blockDim.x + threadIdx.x;
  int total = nChunks * 512;
  if (idx >= total) return;
  int c = idx >> 9;
  int r = idx & 511;
  int q = r >> 4;
  int j = r & 15;
  int k  = c * 32 + ((q >> 4) << 4) + j;
  int co = q & 15;
  float v = 0.f;
  if (co < Cout && k < Cin * KV) {
    int ci = k % Cin;
    int kf = k / Cin;
    int o1 = kf / d234; int rem = kf - o1 * d234;
    int o2 = rem / d34; rem -= o2 * d34;
    int o3 = rem / d4;  int o4 = rem - o3 * d4;
    v = w[(long)(co * Cin + ci) * KV + o1 * s1 + o2 * s2 + o3 * s3 + o4 * s4];
  }
  pack[idx] = (bf16)v;
}

// ------------------------------------------------ symmetric 4D conv (WMMA) ---
// Implicit GEMM, LDS-free, software-pipelined. Channel-last activations make
// the A-operand's two 16B lane segments contiguous in global memory (coalesced
// 512B wave transactions). Each wave owns 2 independent 16x16 output tiles
// (M=32) sharing the B operand. Tap descriptors are wave-uniform and forced
// scalar via readfirstlane (s_load + SALU unpack, branchless cndmask clamp).
// Tables prefetched 2 chunks ahead, A/B operands 1 chunk ahead.
__global__ __launch_bounds__(256) void conv4d_sym_wmma(
    const bf16* __restrict__ xin,         // [B][SP][Cin] channel-last
    const bf16* __restrict__ packA,
    const bf16* __restrict__ packB,
    const int2* __restrict__ ktabA,
    const int2* __restrict__ ktabB,
    const bf16* __restrict__ zrow,        // >=16 zero bf16 for OOB clamp
    const float* __restrict__ bias,
    void* __restrict__ outp, int outIsF32,
    int Cin, int Cout, int coBase, int Ctot,
    int chunksA, int chunksB) {
  const int tid  = threadIdx.x;
  const int lane = tid & 31;
  const int wave = tid >> 5;
  const int g    = lane >> 4;
  const int mw   = lane & 15;

  const long mbase = (long)blockIdx.x * 256;     // 8 waves x 32 rows
  const int nb   = (int)(mbase / SP);
  const int mloc = (int)(mbase % SP);

  // per-tile output-row coordinates (compile-time-constant divisors)
  int h1[2], w1c[2], h2[2], w2c[2];
  const bf16* xrow[2];
  const bf16* xrowg[2];
  long zoff[2];                                  // zero-row clamp as an offset
#pragma unroll
  for (int t = 0; t < 2; ++t) {
    const int m = mloc + wave * 32 + t * 16 + mw;
    const int a = m / S3;    int rm = m - a * S3;
    const int b = rm / S2;   rm -= b * S2;
    const int c = rm / SDIM; const int d = rm - c * SDIM;
    h1[t] = a; w1c[t] = b; h2[t] = c; w2c[t] = d;
    xrow[t]  = xin + ((long)nb * SP + m) * Cin;
    xrowg[t] = xrow[t] + g * 8;
    zoff[t]  = (zrow + g * 8) - xrowg[t];
  }

  const int n = lane & 15;
  const float bn = (n < Cout) ? bias[n] : 0.f;

  const bf16* packs[2] = {packA, packB};
  const int2* ktabs[2] = {ktabA, ktabB};
  const int   nch[2]   = {chunksA, chunksB};

  v8f res[2];

#pragma unroll
  for (int ps = 0; ps < 2; ++ps) {
    const bf16* pk = packs[ps];
    const int2* kt = ktabs[ps];
    const int chunks = nch[ps];
    v8f acc0 = {}; v8f acc1 = {};

    auto loadB = [&](int ch) -> v16bf {
      const bf16* bp = pk + (long)ch * 512 + (lane & 15) * 16 + (lane >> 4) * 8;
      v8bf blo = *(const v8bf*)bp;
      v8bf bhi = *(const v8bf*)(bp + 256);
      return __builtin_shufflevector(blo, bhi,
               0,1,2,3,4,5,6,7,8,9,10,11,12,13,14,15);
    };

    if (Cin == 16) {
      // gather one 16B A segment for tile t under a wave-uniform tap.
      // readfirstlane gives both fields an unconditional scalar use: the table
      // load cannot be sunk into the ok-branch, and the clamp is a cndmask.
      auto gatherSeg = [&](int2 E, int t) -> v8bf {
        const int dly = __builtin_amdgcn_readfirstlane(E.x);
        const int pky = __builtin_amdgcn_readfirstlane(E.y);
        const int c1 = h1[t]  + ((pky << 24) >> 24);
        const int c2 = w1c[t] + ((pky << 16) >> 24);
        const int c3 = h2[t]  + ((pky <<  8) >> 24);
        const int c4 = w2c[t] + ( pky        >> 24);
        const bool ok = ((unsigned)c1 < SDIM) && ((unsigned)c2 < SDIM) &&
                        ((unsigned)c3 < SDIM) && ((unsigned)c4 < SDIM);
        const long o = ok ? ((long)dly * 16) : zoff[t];
        return *(const v8bf*)(xrowg[t] + o);
      };

      // -------- pipelined main path --------
      int2 ec0 = kt[0], ec1 = kt[1];                 // taps for chunk 0
      v16bf A0 = __builtin_shufflevector(gatherSeg(ec0, 0), gatherSeg(ec1, 0),
                   0,1,2,3,4,5,6,7,8,9,10,11,12,13,14,15);
      v16bf A1 = __builtin_shufflevector(gatherSeg(ec0, 1), gatherSeg(ec1, 1),
                   0,1,2,3,4,5,6,7,8,9,10,11,12,13,14,15);
      v16bf Bm = loadB(0);
      int2 en0 = kt[2], en1 = kt[3];                 // taps for chunk 1

      for (int ch = 0; ch < chunks; ++ch) {
        // prefetch taps for chunk ch+2 (tables padded: always in-bounds)
        const int2 ef0 = kt[2 * ch + 4];
        const int2 ef1 = kt[2 * ch + 5];
        // stage operands for chunk ch+1 (packs padded by one chunk)
        v16bf nA0 = __builtin_shufflevector(gatherSeg(en0, 0), gatherSeg(en1, 0),
                      0,1,2,3,4,5,6,7,8,9,10,11,12,13,14,15);
        v16bf nA1 = __builtin_shufflevector(gatherSeg(en0, 1), gatherSeg(en1, 1),
                      0,1,2,3,4,5,6,7,8,9,10,11,12,13,14,15);
        v16bf nB  = loadB(ch + 1);
        // compute chunk ch
        acc0 = __builtin_amdgcn_wmma_f32_16x16x32_bf16(false, A0, false, Bm,
                                                       (short)0, acc0, false, false);
        acc1 = __builtin_amdgcn_wmma_f32_16x16x32_bf16(false, A1, false, Bm,
                                                       (short)0, acc1, false, false);
        A0 = nA0; A1 = nA1; Bm = nB;
        en0 = ef0; en1 = ef1;
      }
    } else {
      // -------- Cin == 1 (first layer only, K = 625): scalar gathers --------
      for (int ch = 0; ch < chunks; ++ch) {
        v16bf Av[2];
#pragma unroll
        for (int t = 0; t < 2; ++t) {
          v16bf a;
#pragma unroll
          for (int j = 0; j < 16; ++j) {
            const int K = (j < 8) ? (g * 8 + j) : (16 + g * 8 + (j - 8));
            const int2 e = kt[ch * 32 + K];
            const int c1 = h1[t]  + ((e.y << 24) >> 24);
            const int c2 = w1c[t] + ((e.y << 16) >> 24);
            const int c3 = h2[t]  + ((e.y <<  8) >> 24);
            const int c4 = w2c[t] + ( e.y        >> 24);
            const bool ok = ((unsigned)c1 < SDIM) && ((unsigned)c2 < SDIM) &&
                            ((unsigned)c3 < SDIM) && ((unsigned)c4 < SDIM);
            const bf16* p = ok ? (xrow[t] + e.x) : zrow;
            a[j] = *p;
          }
          Av[t] = a;
        }
        v16bf Bm = loadB(ch);
        acc0 = __builtin_amdgcn_wmma_f32_16x16x32_bf16(false, Av[0], false, Bm,
                                                       (short)0, acc0, false, false);
        acc1 = __builtin_amdgcn_wmma_f32_16x16x32_bf16(false, Av[1], false, Bm,
                                                       (short)0, acc1, false, false);
      }
    }

    if (ps == 0) {
#pragma unroll
      for (int r = 0; r < 8; ++r) { res[0][r] = fmaxf(acc0[r] + bn, 0.f);
                                    res[1][r] = fmaxf(acc1[r] + bn, 0.f); }
    } else {
#pragma unroll
      for (int r = 0; r < 8; ++r) { res[0][r] += fmaxf(acc0[r] + bn, 0.f);
                                    res[1][r] += fmaxf(acc1[r] + bn, 0.f); }
    }
  }

  // epilogue: D layout lane -> N = lane&15, M = g*8 + r (per 16x16 tile)
  if (n < Cout) {
    const int co = coBase + n;
#pragma unroll
    for (int t = 0; t < 2; ++t) {
      const int mOut = mloc + wave * 32 + t * 16 + g * 8;
      if (outIsF32) {
        float* o = (float*)outp + ((long)nb * Ctot + co) * SP + mOut;
#pragma unroll
        for (int r = 0; r < 8; ++r) o[r] = res[t][r];
      } else {
        bf16* o = (bf16*)outp + ((long)nb * SP + mOut) * (long)Ctot + co;
#pragma unroll
        for (int r = 0; r < 8; ++r) o[(long)r * Ctot] = (bf16)res[t][r];
      }
    }
  }
}

// -------------------------------------------------------------------- host ---

static inline int ceil_div_i(int a, int b) { return (a + b - 1) / b; }

extern "C" void kernel_launch(void* const* d_in, const int* in_sizes, int n_in,
                              void* d_out, int out_size, void* d_ws, size_t ws_size,
                              hipStream_t stream) {
  (void)in_sizes; (void)n_in; (void)out_size; (void)ws_size;
  const float* x   = (const float*)d_in[0];
  const float* w0  = (const float*)d_in[1];
  const float* b0  = (const float*)d_in[2];
  const float* w10 = (const float*)d_in[3];
  const float* b10 = (const float*)d_in[4];
  const float* w11 = (const float*)d_in[5];
  const float* b11 = (const float*)d_in[6];
  const float* w2  = (const float*)d_in[7];
  const float* b2  = (const float*)d_in[8];
  float* out = (float*)d_out;

  const int B = 4;
  size_t off = 0;
  auto take = [&](size_t bytes) -> void* {
    off = (off + 255) & ~(size_t)255;
    void* p = (char*)d_ws + off;
    off += bytes;
    return p;
  };

  bf16* xin  = (bf16*)take((size_t)B * SP * 2);            // [B][SP][1]
  bf16* buf1 = (bf16*)take((size_t)B * SP * 16 * 2);       // [B][SP][16]
  bf16* buf2 = (bf16*)take((size_t)B * SP * 16 * 2);       // [B][SP][16]
  bf16* zrow = (bf16*)take(64);                            // zero row for OOB

  const int ch0  = ceil_div_i(1  * 625, 32);   // 20
  const int ch10 = ceil_div_i(16 * 225, 32);   // 113
  const int ch11 = ceil_div_i(16 * 625, 32);   // 313
  const int ch2  = ceil_div_i(16 * 625, 32);   // 313

  // packs padded by one chunk for the pipeline's one-ahead B prefetch
  bf16* p0a  = (bf16*)take((size_t)(ch0  + 1) * 512 * 2);
  bf16* p0b  = (bf16*)take((size_t)(ch0  + 1) * 512 * 2);
  bf16* p10a = (bf16*)take((size_t)(ch10 + 1) * 512 * 2);
  bf16* p10b = (bf16*)take((size_t)(ch10 + 1) * 512 * 2);
  bf16* p11a = (bf16*)take((size_t)(ch11 + 1) * 512 * 2);
  bf16* p11b = (bf16*)take((size_t)(ch11 + 1) * 512 * 2);
  bf16* p2a  = (bf16*)take((size_t)(ch2  + 1) * 512 * 2);
  bf16* p2b  = (bf16*)take((size_t)(ch2  + 1) * 512 * 2);

  // tables padded +4 entries for the two-ahead tap prefetch
  const int kf0 = ch0 * 32 + 4, kf10 = ch10 * 2 + 4;
  const int kf11 = ch11 * 2 + 4, kf2 = ch2 * 2 + 4;
  int2* t0a  = (int2*)take((size_t)kf0  * sizeof(int2));
  int2* t0b  = (int2*)take((size_t)kf0  * sizeof(int2));
  int2* t10a = (int2*)take((size_t)kf10 * sizeof(int2));
  int2* t10b = (int2*)take((size_t)kf10 * sizeof(int2));
  int2* t11a = (int2*)take((size_t)kf11 * sizeof(int2));
  int2* t11b = (int2*)take((size_t)kf11 * sizeof(int2));
  int2* t2a  = (int2*)take((size_t)kf2  * sizeof(int2));
  int2* t2b  = (int2*)take((size_t)kf2  * sizeof(int2));

  { // input fp32 -> bf16, and the zero row
    int n = B * SP;
    cvt_f32_to_bf16<<<ceil_div_i(n, 256), 256, 0, stream>>>(x, xin, n);
    zero_bf16<<<1, 64, 0, stream>>>(zrow, 32);
  }

  auto prep = [&](const float* w, bf16* pa, bf16* pb, int2* ta, int2* tb,
                  int Cout, int Cin, int k1, int k2, int k3, int k4,
                  int chunks, int maxKf) {
    int KV = k1 * k2 * k3 * k4;
    int blocks = ceil_div_i(chunks * 512, 256);
    // normal: decompose over (k1,k2,k3,k4), strides (k2k3k4, k3k4, k4, 1)
    repack_weights<<<blocks, 256, 0, stream>>>(w, pa, Cout, Cin, KV,
        k4, k3 * k4, k2 * k3 * k4,
        k2 * k3 * k4, k3 * k4, k4, 1, chunks);
    // permuted kernel: decompose over (k3,k4,k1,k2), strides (k4, 1, k2k3k4, k3k4)
    repack_weights<<<blocks, 256, 0, stream>>>(w, pb, Cout, Cin, KV,
        k2, k1 * k2, k4 * k1 * k2,
        k4, 1, k2 * k3 * k4, k3 * k4, chunks);
    int tblk = ceil_div_i(maxKf, 256);
    build_ktab<<<tblk, 256, 0, stream>>>(ta, maxKf, k1, k2, k3, k4);
    build_ktab<<<tblk, 256, 0, stream>>>(tb, maxKf, k3, k4, k1, k2);
  };

  prep(w0,  p0a,  p0b,  t0a,  t0b, 16,  1, 5, 5, 5, 5, ch0,  kf0);
  prep(w10, p10a, p10b, t10a, t10b, 8, 16, 3, 3, 5, 5, ch10, kf10);
  prep(w11, p11a, p11b, t11a, t11b, 8, 16, 5, 5, 5, 5, ch11, kf11);
  prep(w2,  p2a,  p2b,  t2a,  t2b,  1, 16, 5, 5, 5, 5, ch2,  kf2);

  const int convBlocks = B * SP / 256;  // 2500 (8 waves x 2 tiles x 16 rows)

  // layer 0: Cin=1 -> 16 channels
  conv4d_sym_wmma<<<convBlocks, 256, 0, stream>>>(xin, p0a, p0b, t0a, t0b, zrow,
      b0, buf1, 0, 1, 16, 0, 16, ch0, ch0);
  // layer 1 branch 0: 16 -> 8 (channels [0,8))
  conv4d_sym_wmma<<<convBlocks, 256, 0, stream>>>(buf1, p10a, p10b, t10a, t10b, zrow,
      b10, buf2, 0, 16, 8, 0, 16, ch10, ch10);
  // layer 1 branch 1: 16 -> 8 (channels [8,16))
  conv4d_sym_wmma<<<convBlocks, 256, 0, stream>>>(buf1, p11a, p11b, t11a, t11b, zrow,
      b11, buf2, 0, 16, 8, 8, 16, ch11, ch11);
  // layer 2: 16 -> 1, fp32 output
  conv4d_sym_wmma<<<convBlocks, 256, 0, stream>>>(buf2, p2a, p2b, t2a, t2b, zrow,
      b2, out, 1, 16, 1, 0, 1, ch2, ch2);
}
